// GraphSAGE_Encoder_69947837382766
// MI455X (gfx1250) — matrix operand order repaired and checked by reading, not compiled, not run on hardware
//
#include <hip/hip_runtime.h>
#include <hip/hip_bf16.h>

typedef __attribute__((ext_vector_type(16))) _Float16 v16h;
typedef __attribute__((ext_vector_type(8)))  float    v8f;

// ---------------------------------------------------------------------------
// Fast f32 atomic add: relaxed / agent scope lowers to global_atomic_add_f32
// ---------------------------------------------------------------------------
__device__ __forceinline__ void atomAddF(float* p, float v) {
    __hip_atomic_fetch_add(p, v, __ATOMIC_RELAXED, __HIP_MEMORY_SCOPE_AGENT);
}

// ---------------------------------------------------------------------------
// f32 -> f16 weight conversion (64x64 matrices)
// ---------------------------------------------------------------------------
__global__ void cvt_f16_kernel(const float* __restrict__ in,
                               _Float16* __restrict__ out, int n) {
    int i = blockIdx.x * blockDim.x + threadIdx.x;
    if (i < n) out[i] = (_Float16)in[i];
}

// ---------------------------------------------------------------------------
// Degree accumulation: one thread per edge
// ---------------------------------------------------------------------------
__global__ void deg_kernel(const long long* __restrict__ dst,
                           float* __restrict__ deg, int E) {
    int e = blockIdx.x * blockDim.x + threadIdx.x;
    if (e < E) atomAddF(&deg[(int)dst[e]], 1.0f);
}

__global__ void invdeg_kernel(float* __restrict__ deg, int n) {
    int i = blockIdx.x * blockDim.x + threadIdx.x;
    if (i < n) deg[i] = 1.0f / fmaxf(deg[i], 1.0f);
}

// ---------------------------------------------------------------------------
// Edge scatter: one wave per edge. Edge index is wave-uniform -> force it
// scalar with readfirstlane so src/dst become s_loads. Lane l reads float2 at
// col 2l of x[src] (coalesced 256B per wave) and atomically adds into
// agg[dst]. x and agg are 25.6MB each -> L2 resident on the 192MB L2, so
// this runs at L2 atomic throughput (the roofline term of the whole net).
// ---------------------------------------------------------------------------
__global__ void scatter_kernel(const float* __restrict__ x,
                               const long long* __restrict__ src,
                               const long long* __restrict__ dst,
                               float* __restrict__ agg, int E) {
    int tid  = blockIdx.x * blockDim.x + threadIdx.x;
    int e    = __builtin_amdgcn_readfirstlane(tid >> 5);  // wave-uniform
    int lane = tid & 31;
    if (e >= E) return;
    int s = (int)src[e];   // scalar load
    int d = (int)dst[e];   // scalar load
    const float2 v = *(const float2*)(x + (size_t)s * 64 + lane * 2);
    float* a = agg + (size_t)d * 64 + lane * 2;
    atomAddF(a,     v.x);
    atomAddF(a + 1, v.y);
}

// ---------------------------------------------------------------------------
// A-fragment loader for v_wmma_f32_16x16x32_f16.
// Lane L holds row M = L%16; element e of the v16h maps to
// K = kstep*32 + (L>=16 ? 16 : 0) + e (same contiguous order as B).
// => one contiguous 16-float chunk at k0 = kstep*32 + hi*16.
// ---------------------------------------------------------------------------
__device__ __forceinline__ v16h load_a_frag(const float* __restrict__ row,
                                            int kstep, int hi, float scale) {
    int b0 = kstep * 32 + hi * 16;
    const float4* p0 = (const float4*)(row + b0);
    const float4* p1 = (const float4*)(row + b0 + 8);
    float4 q0 = p0[0], q1 = p0[1];
    float4 q2 = p1[0], q3 = p1[1];
    float f[16] = {q0.x, q0.y, q0.z, q0.w, q1.x, q1.y, q1.z, q1.w,
                   q2.x, q2.y, q2.z, q2.w, q3.x, q3.y, q3.z, q3.w};
    v16h a;
#pragma unroll
    for (int i = 0; i < 16; i++) a[i] = (_Float16)(f[i] * scale);
    return a;
}

// ---------------------------------------------------------------------------
// Fused SAGE linear:  out = act( (agg*invdeg) @ Wl^T + bias + xin @ Wr^T )
// One wave owns 16 nodes x 64 output features (4 C tiles, 16 WMMAs).
// B layout (16-bit B 32x16, contiguous k order chosen here): lane L holds col N = L%16, element e
// maps to K = kstep*32 + (L>=16 ? 16:0) + e  -> contiguous 16 halves of the
// row-major f16 weight row => single 32B v16h load.
// C/D layout: lane L covers N = L%16, VGPR r covers M = r + 8*(L/16).
// Compile-time flags: USE_AGG=0/RELU=0 gives plain head (xin @ Wr^T + bias).
// All branches that could split EXEC are wave-uniform (WMMA needs EXEC=all-1).
// ---------------------------------------------------------------------------
template <int USE_AGG, int RELU>
__global__ void sage_wmma_kernel(const float* __restrict__ agg,
                                 const float* __restrict__ invdeg,
                                 const float* __restrict__ xin,
                                 const _Float16* __restrict__ Wl,
                                 const _Float16* __restrict__ Wr,
                                 const float* __restrict__ bias,
                                 float* __restrict__ out,
                                 int n_nodes) {
    int tid  = blockIdx.x * blockDim.x + threadIdx.x;
    int tile = __builtin_amdgcn_readfirstlane(tid >> 5);  // wave-uniform tile
    int lane = tid & 31;
    if (tile * 16 >= n_nodes) return;  // scalar-uniform exit: EXEC all-1s kept
    const bool full = (tile * 16 + 16 <= n_nodes);        // wave-uniform
    int row = lane & 15;
    int hi  = lane >> 4;
    int node = tile * 16 + row;
    int nodec = node < n_nodes ? node : (n_nodes - 1);    // clamp loads only

    const float* xrow = xin + (size_t)nodec * 64;
    v16h ax0 = load_a_frag(xrow, 0, hi, 1.0f);
    v16h ax1 = load_a_frag(xrow, 1, hi, 1.0f);
    v16h am0 = ax0, am1 = ax1;
    if (USE_AGG) {
        float inv = invdeg[nodec];
        const float* arow = agg + (size_t)nodec * 64;
        am0 = load_a_frag(arow, 0, hi, inv);
        am1 = load_a_frag(arow, 1, hi, inv);
    }

#pragma unroll
    for (int tn = 0; tn < 4; tn++) {
        int o = tn * 16 + row;          // global output feature for this lane
        v8f c = {};
        if (USE_AGG) {
            v16h b0 = *(const v16h*)(Wl + (size_t)o * 64 + hi * 16);
            v16h b1 = *(const v16h*)(Wl + (size_t)o * 64 + 32 + hi * 16);
            c = __builtin_amdgcn_wmma_f32_16x16x32_f16(false, am0, false, b0,
                                                       (short)0, c, false, false);
            c = __builtin_amdgcn_wmma_f32_16x16x32_f16(false, am1, false, b1,
                                                       (short)0, c, false, false);
        }
        {
            v16h b0 = *(const v16h*)(Wr + (size_t)o * 64 + hi * 16);
            v16h b1 = *(const v16h*)(Wr + (size_t)o * 64 + 32 + hi * 16);
            c = __builtin_amdgcn_wmma_f32_16x16x32_f16(false, ax0, false, b0,
                                                       (short)0, c, false, false);
            c = __builtin_amdgcn_wmma_f32_16x16x32_f16(false, ax1, false, b1,
                                                       (short)0, c, false, false);
        }
        float bv = bias[o];
        // single base pointer; 8 stores with immediate offsets (r*256 bytes)
        float* po = out + (size_t)(tile * 16 + 8 * hi) * 64 + o;
        if (full) {
#pragma unroll
            for (int r = 0; r < 8; r++) {
                float v = c[r] + bv;
                if (RELU) v = fmaxf(v, 0.0f);
                po[(size_t)r * 64] = v;
            }
        } else {
#pragma unroll
            for (int r = 0; r < 8; r++) {
                float v = c[r] + bv;
                if (RELU) v = fmaxf(v, 0.0f);
                if (tile * 16 + 8 * hi + r < n_nodes) po[(size_t)r * 64] = v;
            }
        }
    }
}

// ---------------------------------------------------------------------------
// In-place log_softmax over rows of 64: one wave per node, shfl reductions.
// ---------------------------------------------------------------------------
__global__ void log_softmax_kernel(float* __restrict__ out, int n_nodes) {
    int tid  = blockIdx.x * blockDim.x + threadIdx.x;
    int w    = tid >> 5;
    int lane = tid & 31;
    if (w >= n_nodes) return;
    float* row = out + (size_t)w * 64;
    float v0 = row[lane], v1 = row[lane + 32];
    float m = fmaxf(v0, v1);
#pragma unroll
    for (int off = 16; off > 0; off >>= 1)
        m = fmaxf(m, __shfl_xor(m, off, 32));
    float s = __expf(v0 - m) + __expf(v1 - m);
#pragma unroll
    for (int off = 16; off > 0; off >>= 1)
        s += __shfl_xor(s, off, 32);
    float lse = m + __logf(s);
    row[lane]      = v0 - lse;
    row[lane + 32] = v1 - lse;
}

// ---------------------------------------------------------------------------
// Launch: deg -> invdeg -> [scatter, sage] x2 -> head gemm -> log_softmax.
// Only stream-ordered ops (graph-capture safe).
// ---------------------------------------------------------------------------
extern "C" void kernel_launch(void* const* d_in, const int* in_sizes, int n_in,
                              void* d_out, int out_size, void* d_ws, size_t ws_size,
                              hipStream_t stream) {
    const float*     x    = (const float*)d_in[0];
    const long long* ei   = (const long long*)d_in[1];
    const float*     W1l  = (const float*)d_in[2];
    const float*     b1l  = (const float*)d_in[3];
    const float*     W1r  = (const float*)d_in[4];
    const float*     W2l  = (const float*)d_in[5];
    const float*     b2l  = (const float*)d_in[6];
    const float*     W2r  = (const float*)d_in[7];
    const float*     Wout = (const float*)d_in[8];
    const float*     bout = (const float*)d_in[9];

    const int N = in_sizes[0] / 64;
    const int E = in_sizes[1] / 2;
    const long long* src = ei;
    const long long* dst = ei + E;

    // workspace carve-out (256B aligned)
    char*  ws  = (char*)d_ws;
    size_t off = 0;
    auto carve = [&](size_t bytes) -> void* {
        void* p = ws + off;
        off += (bytes + 255) & ~(size_t)255;
        return p;
    };
    float*    invdeg = (float*)carve((size_t)N * 4);
    float*    aggbuf = (float*)carve((size_t)N * 64 * 4);
    float*    h1     = (float*)carve((size_t)N * 64 * 4);
    float*    h2     = (float*)carve((size_t)N * 64 * 4);
    _Float16* Wh     = (_Float16*)carve(5 * 4096 * sizeof(_Float16));
    _Float16 *W1l_h = Wh,            *W1r_h = Wh + 4096,
             *W2l_h = Wh + 2 * 4096, *W2r_h = Wh + 3 * 4096,
             *Wout_h = Wh + 4 * 4096;

    const int TB = 256;
    const int wblk   = 16;                      // weight cvt blocks: 4096/256
    const int eblk   = (E + TB - 1) / TB;
    const int nblk   = (N + TB - 1) / TB;
    const int sblk   = (int)(((size_t)E * 32 + TB - 1) / TB);
    const int tiles  = (N + 15) / 16;
    const int gblk   = (int)(((size_t)tiles * 32 + TB - 1) / TB);
    const int smblk  = (int)(((size_t)N * 32 + TB - 1) / TB);

    // init accumulators (must re-zero every call: deterministic, no state)
    hipMemsetAsync(invdeg, 0, (size_t)N * 4, stream);
    hipMemsetAsync(aggbuf, 0, (size_t)N * 64 * 4, stream);

    // weights -> f16
    cvt_f16_kernel<<<wblk, TB, 0, stream>>>(W1l,  W1l_h,  4096);
    cvt_f16_kernel<<<wblk, TB, 0, stream>>>(W1r,  W1r_h,  4096);
    cvt_f16_kernel<<<wblk, TB, 0, stream>>>(W2l,  W2l_h,  4096);
    cvt_f16_kernel<<<wblk, TB, 0, stream>>>(W2r,  W2r_h,  4096);
    cvt_f16_kernel<<<wblk, TB, 0, stream>>>(Wout, Wout_h, 4096);

    // degrees (shared by both layers)
    deg_kernel<<<eblk, TB, 0, stream>>>(dst, invdeg, E);
    invdeg_kernel<<<nblk, TB, 0, stream>>>(invdeg, N);

    // layer 1
    scatter_kernel<<<sblk, TB, 0, stream>>>(x, src, dst, aggbuf, E);
    sage_wmma_kernel<1, 1><<<gblk, TB, 0, stream>>>(aggbuf, invdeg, x,
                                                    W1l_h, W1r_h, b1l, h1, N);

    // layer 2
    hipMemsetAsync(aggbuf, 0, (size_t)N * 64 * 4, stream);
    scatter_kernel<<<sblk, TB, 0, stream>>>(h1, src, dst, aggbuf, E);
    sage_wmma_kernel<1, 1><<<gblk, TB, 0, stream>>>(aggbuf, invdeg, h1,
                                                    W2l_h, W2r_h, b2l, h2, N);

    // output head: logits = h2 @ Wout^T + bout
    sage_wmma_kernel<0, 0><<<gblk, TB, 0, stream>>>(aggbuf, invdeg, h2,
                                                    Wout_h, Wout_h, bout,
                                                    (float*)d_out, N);

    // log_softmax in place on d_out
    log_softmax_kernel<<<smblk, TB, 0, stream>>>((float*)d_out, N);
}
